// dis_flow_27951647162956
// MI455X (gfx1250) — compile-verified, matrix operand order checked by reading, not run on hardware
//
#include <hip/hip_runtime.h>
#include <hip/hip_bf16.h>

typedef __bf16 bf16_t;
typedef __attribute__((ext_vector_type(16))) __bf16 v16bf;
typedef __attribute__((ext_vector_type(8)))  float  v8f;
typedef int v4i __attribute__((vector_size(16)));
typedef __attribute__((address_space(1))) v4i* gv4i_ptr;  // global (__device__)
typedef __attribute__((address_space(3))) v4i* lv4i_ptr;  // LDS (__shared__)

#if defined(__has_builtin)
#if __has_builtin(__builtin_amdgcn_global_load_async_to_lds_b128) && \
    __has_builtin(__builtin_amdgcn_s_wait_asynccnt)
#define USE_ASYNC_LDS 1
#endif
#endif

namespace {
constexpr int B_ROWS = 16384;
constexpr int UNIT   = 128;
constexpr int VOC    = 11;          // one-hot width per unit
constexpr int D_IN   = UNIT * VOC;  // 1408
constexpr int H_DIM  = 4096;

constexpr int TILE_M = 256;
constexpr int TILE_N = 128;
constexpr int TILE_K = 32;
constexpr int LDSS   = TILE_K + 8;  // padded bf16 row stride (80B, 16B aligned)
}

union FragU { uint4 q[2]; v16bf v; };

// ---------------------------------------------------------------- convert X
__global__ void cvt_f32_bf16(const float* __restrict__ src,
                             bf16_t* __restrict__ dst, long n) {
  long i = ((long)blockIdx.x * blockDim.x + threadIdx.x) * 4;
  if (i >= n) return;
  float4 f = *reinterpret_cast<const float4*>(src + i);
  dst[i + 0] = (bf16_t)f.x;
  dst[i + 1] = (bf16_t)f.y;
  dst[i + 2] = (bf16_t)f.z;
  dst[i + 3] = (bf16_t)f.w;
}

// ------------------------------------------------- mask + pack weights bf16
// mode 0: m1[n,k]  = (n%127) >= (k/11)     (W1:  H x D)
// mode 1: m2[n,k]  = (n%127) >= (k%127)    (W2:  H x H)
// mode 2: mout[n,k]= (n/11)  >  (k%127)    (Wout:D x H)
__global__ void pack_w_kernel(const float* __restrict__ W,
                              bf16_t* __restrict__ Wt,
                              int K, int mode, long total) {
  long i = ((long)blockIdx.x * blockDim.x + threadIdx.x) * 4;
  if (i >= total) return;
#pragma unroll
  for (int j = 0; j < 4; ++j) {
    long idx = i + j;
    int n = (int)(idx / K);
    int k = (int)(idx % K);
    bool keep;
    if (mode == 0)      keep = (n % 127) >= (k / VOC);
    else if (mode == 1) keep = (n % 127) >= (k % 127);
    else                keep = (n / VOC) > (k % 127);
    float v = keep ? W[idx] : 0.0f;
    Wt[idx] = (bf16_t)v;
  }
}

// --------------------------------------------------------- tiled WMMA GEMM
// C[m,n] = act( sum_k A[m,k] * Bt[n,k] + bias[n] )
// A: M x K bf16 row-major, Bt: N x K bf16 row-major.
// Block tile 256x128, 8 waves (4 along M x 2 along N), 64x64 per wave
// (16 WMMAs / K-step).
// Preferred path: double-buffered LDS fed by GLOBAL_LOAD_ASYNC_TO_LDS_B128
// (ASYNCcnt) -> zero staging VGPRs, one barrier per K-step.
// Fallback: short-lived register staging (no cross-barrier live range).
template <bool RELU, bool BF16OUT>
__global__ __launch_bounds__(256, 1) void gemm_bf16_wmma(
    const bf16_t* __restrict__ A, const bf16_t* __restrict__ Bt,
    const float* __restrict__ bias, void* __restrict__ Cout,
    int N, int K) {
  __shared__ bf16_t As[2][TILE_M * LDSS];  // 2 x 256 x 40 bf16 = 40 KB
  __shared__ bf16_t Bs[2][TILE_N * LDSS];  // 2 x 128 x 40 bf16 = 20 KB

  const int tid  = threadIdx.x;
  const int lane = tid & 31;
  const int wave = tid >> 5;
  const int wm   = (wave & 3) * 64;   // 4 waves along M
  const int wn   = (wave >> 2) * 64;  // 2 waves along N
  const int half = lane >> 4;         // hi/lo 16-lane group
  const int l16  = lane & 15;

  const long row0 = (long)blockIdx.y * TILE_M;
  const int  col0 = blockIdx.x * TILE_N;

  // per-thread cooperative-load coordinates (16B chunks of a 32-wide K slab)
  const bf16_t* ap[4];
  const bf16_t* bp[2];
  int asoff[4], bsoff[2];
#pragma unroll
  for (int i = 0; i < 4; ++i) {
    int c = tid + i * 256;          // 0..1023 -> 256 rows x 4 chunks
    int r = c >> 2, cc = (c & 3) << 3;
    ap[i] = A + (row0 + r) * K + cc;
    asoff[i] = r * LDSS + cc;
  }
#pragma unroll
  for (int i = 0; i < 2; ++i) {
    int c = tid + i * 256;          // 0..511 -> 128 rows x 4 chunks
    int r = c >> 2, cc = (c & 3) << 3;
    bp[i] = Bt + (long)(col0 + r) * K + cc;
    bsoff[i] = r * LDSS + cc;
  }

  v8f acc[4][4] = {};
  const int nk = K / TILE_K;

  // 16-bit 16x32 operand layout: lanes 0-15 hold K {0..7,16..23},
  // lanes 16-31 hold K {8..15,24..31} -> two 16B LDS reads per fragment.
  auto compute = [&](int cur) {
    v16bf bfrag[4];
#pragma unroll
    for (int ni = 0; ni < 4; ++ni) {
      FragU f;
      const bf16_t* p = &Bs[cur][(wn + ni * 16 + l16) * LDSS];
      f.q[0] = *reinterpret_cast<const uint4*>(p + half * 8);
      f.q[1] = *reinterpret_cast<const uint4*>(p + 16 + half * 8);
      bfrag[ni] = f.v;
    }
#pragma unroll
    for (int mi = 0; mi < 4; ++mi) {
      FragU f;
      const bf16_t* p = &As[cur][(wm + mi * 16 + l16) * LDSS];
      f.q[0] = *reinterpret_cast<const uint4*>(p + half * 8);
      f.q[1] = *reinterpret_cast<const uint4*>(p + 16 + half * 8);
      v16bf afrag = f.v;
#pragma unroll
      for (int ni = 0; ni < 4; ++ni)
        acc[mi][ni] = __builtin_amdgcn_wmma_f32_16x16x32_bf16(
            false, afrag, false, bfrag[ni], (short)0, acc[mi][ni],
            false, false);
    }
  };

#ifdef USE_ASYNC_LDS
  // ---- async global->LDS DMA double buffer, ASYNCcnt tracked ----
  auto issue_tile = [&](int buf) {
#pragma unroll
    for (int i = 0; i < 4; ++i) {
      __builtin_amdgcn_global_load_async_to_lds_b128(
          (gv4i_ptr)(void*)const_cast<bf16_t*>(ap[i]),
          (lv4i_ptr)(void*)&As[buf][asoff[i]],
          0, 0);
      ap[i] += TILE_K;
    }
#pragma unroll
    for (int i = 0; i < 2; ++i) {
      __builtin_amdgcn_global_load_async_to_lds_b128(
          (gv4i_ptr)(void*)const_cast<bf16_t*>(bp[i]),
          (lv4i_ptr)(void*)&Bs[buf][bsoff[i]],
          0, 0);
      bp[i] += TILE_K;
    }
  };

  issue_tile(0);
  for (int kt = 0; kt < nk; ++kt) {
    const int cur = kt & 1;
    __builtin_amdgcn_s_wait_asynccnt(0);  // this wave's DMA into LDS[cur] done
    __syncthreads();                      // everyone's DMA done
    if (kt + 1 < nk) issue_tile(cur ^ 1); // DMA next tile while we compute
    compute(cur);
    // next write to LDS[cur] is after the next iteration's barrier;
    // each wave drains its frag ds_loads (s_wait_dscnt before WMMA) before
    // signaling, so DMA writes cannot overtake prior reads.
  }
#else
  // ---- fallback: short-lived register staging, two barriers per step ----
  for (int kt = 0; kt < nk; ++kt) {
    const int cur = kt & 1;
#pragma unroll
    for (int i = 0; i < 4; ++i) {
      uint4 t = *reinterpret_cast<const uint4*>(ap[i]);
      ap[i] += TILE_K;
      *reinterpret_cast<uint4*>(&As[cur][asoff[i]]) = t;
    }
#pragma unroll
    for (int i = 0; i < 2; ++i) {
      uint4 t = *reinterpret_cast<const uint4*>(bp[i]);
      bp[i] += TILE_K;
      *reinterpret_cast<uint4*>(&Bs[cur][bsoff[i]]) = t;
    }
    __syncthreads();
    compute(cur);
    __syncthreads();
  }
#endif

  // epilogue: C frag layout: VGPR r -> M = r + half*8, N = l16
#pragma unroll
  for (int ni = 0; ni < 4; ++ni) {
    int n = col0 + wn + ni * 16 + l16;
    float bv = bias[n];
#pragma unroll
    for (int mi = 0; mi < 4; ++mi) {
#pragma unroll
      for (int r = 0; r < 8; ++r) {
        long m = row0 + wm + mi * 16 + half * 8 + r;
        float v = acc[mi][ni][r] + bv;
        if (RELU) v = fmaxf(v, 0.0f);
        if (BF16OUT)
          reinterpret_cast<bf16_t*>(Cout)[m * N + n] = (bf16_t)v;
        else
          reinterpret_cast<float*>(Cout)[m * N + n] = v;
      }
    }
  }
}

// ---------------------------------------------- argmax + modular shift tail
// out[b,u] = one_hot((argmax_v x[b,u,:] + argmax_v logits[b,u,:]) mod 11)
__global__ void onehot_shift_kernel(const float* __restrict__ x,
                                    const float* __restrict__ lg,
                                    float* __restrict__ out, int total) {
  int i = blockIdx.x * blockDim.x + threadIdx.x;
  if (i >= total) return;
  long base = (long)i * VOC;
  float best = lg[base];
  int c = 0;
#pragma unroll
  for (int v = 1; v < VOC; ++v) {
    float t = lg[base + v];
    if (t > best) { best = t; c = v; }
  }
  int a = 0;
#pragma unroll
  for (int v = 0; v < VOC; ++v)
    if (x[base + v] > 0.5f) { a = v; break; }
  int o = a + c;
  if (o >= VOC) o -= VOC;
#pragma unroll
  for (int v = 0; v < VOC; ++v) out[base + v] = (v == o) ? 1.0f : 0.0f;
}

// --------------------------------------------------------------- launcher
extern "C" void kernel_launch(void* const* d_in, const int* in_sizes, int n_in,
                              void* d_out, int out_size, void* d_ws,
                              size_t ws_size, hipStream_t stream) {
  (void)in_sizes; (void)n_in; (void)out_size; (void)ws_size;
  const float* x    = (const float*)d_in[0];
  const float* W1   = (const float*)d_in[1];
  const float* b1   = (const float*)d_in[2];
  const float* W2   = (const float*)d_in[3];
  const float* b2   = (const float*)d_in[4];
  const float* Wout = (const float*)d_in[5];
  const float* bout = (const float*)d_in[6];
  // d_in[7..9] = MADE masks: recomputed analytically in pack_w_kernel.

  const long nX  = (long)B_ROWS * D_IN;   // 23,068,672
  const long nW1 = (long)H_DIM * D_IN;    //  5,767,168
  const long nW2 = (long)H_DIM * H_DIM;   // 16,777,216
  const long nW3 = (long)D_IN * H_DIM;    //  5,767,168
  const long nH  = (long)B_ROWS * H_DIM;  // 67,108,864

  char* ws = (char*)d_ws;
  size_t off = 0;
  bf16_t* Xb  = (bf16_t*)(ws + off); off += (size_t)nX  * 2;
  bf16_t* W1t = (bf16_t*)(ws + off); off += (size_t)nW1 * 2;
  bf16_t* W2t = (bf16_t*)(ws + off); off += (size_t)nW2 * 2;
  bf16_t* W3t = (bf16_t*)(ws + off); off += (size_t)nW3 * 2;
  bf16_t* H1  = (bf16_t*)(ws + off); off += (size_t)nH  * 2;
  bf16_t* H2  = (bf16_t*)(ws + off); off += (size_t)nH  * 2;
  float*  LG  = (float*)H1;  // logits (92MB) reuse H1 region (134MB), free after GEMM2

  // --- pack / convert ---
  cvt_f32_bf16<<<dim3((unsigned)(nX / 4 / 256)), 256, 0, stream>>>(x, Xb, nX);
  pack_w_kernel<<<dim3((unsigned)(nW1 / 4 / 256)), 256, 0, stream>>>(W1, W1t, D_IN, 0, nW1);
  pack_w_kernel<<<dim3((unsigned)(nW2 / 4 / 256)), 256, 0, stream>>>(W2, W2t, H_DIM, 1, nW2);
  pack_w_kernel<<<dim3((unsigned)(nW3 / 4 / 256)), 256, 0, stream>>>(Wout, W3t, H_DIM, 2, nW3);

  // --- 3 GEMMs ---
  dim3 blk(256);
  dim3 g1(H_DIM / TILE_N, B_ROWS / TILE_M);  // 32 x 64
  gemm_bf16_wmma<true, true><<<g1, blk, 0, stream>>>(Xb, W1t, b1, H1, H_DIM, D_IN);
  gemm_bf16_wmma<true, true><<<g1, blk, 0, stream>>>(H1, W2t, b2, H2, H_DIM, H_DIM);
  dim3 g3(D_IN / TILE_N, B_ROWS / TILE_M);   // 11 x 64
  gemm_bf16_wmma<false, false><<<g3, blk, 0, stream>>>(H2, W3t, bout, LG, D_IN, H_DIM);

  // --- argmax + circular shift ---
  int total = B_ROWS * UNIT;  // 2,097,152
  onehot_shift_kernel<<<dim3(total / 256), 256, 0, stream>>>(x, LG, (float*)d_out, total);
}
